// MSSAttention_73005854097762
// MI455X (gfx1250) — compile-verified, matrix operand order checked by reading, not run on hardware
//
#include <hip/hip_runtime.h>
#include <math.h>

typedef __attribute__((ext_vector_type(16))) _Float16 v16h;
typedef __attribute__((ext_vector_type(8)))  float    v8f;

// ---- problem constants ----
constexpr int kB = 8, kC = 256, kH = 56, kW = 56, kHW = kH * kW;
constexpr int kHeads = 8, kHD = 32, kGroup = 8, kCPG = kC / kGroup; // 32
constexpr int kWin = 7, kHP = 8, kWP = 8, kL = kHP * kWP;           // 64
constexpr float kEps = 1e-5f;

// =======================================================================
// CDNA5 async global->LDS staging (ASYNCcnt-tracked DMA, no VGPR roundtrip)
// =======================================================================
__device__ __forceinline__ unsigned lds_off_of(const void* shared_ptr) {
  // addrspacecast AS(3)->generic keeps the LDS byte offset in the low 32 bits
  return (unsigned)(size_t)shared_ptr;
}
__device__ __forceinline__ void async_load_b128(unsigned lds_off, const float* gptr) {
  unsigned long long addr = (unsigned long long)(size_t)gptr;
  asm volatile("global_load_async_to_lds_b128 %0, %1, off"
               :: "v"(lds_off), "v"(addr) : "memory");
}
__device__ __forceinline__ void wait_async0() {
  asm volatile("s_wait_asynccnt 0x0" ::: "memory");
}

// =======================================================================
// WMMA fragment helpers (CDNA5 16x16x32 f16 layouts, wave32)
// =======================================================================
__device__ __forceinline__ int frag_kidx(int e, int lane) {
  return ((e >> 3) << 4) + ((lane >> 4) << 3) + (e & 7);
}

__device__ __forceinline__ v16h load_frag_rowmaj(const _Float16* M, int ld,
                                                 int m0, int k0, int lane) {
  union { v16h v; _Float16 h[16]; } u;
  int row = m0 + (lane & 15);
#pragma unroll
  for (int e = 0; e < 16; ++e) u.h[e] = M[row * ld + k0 + frag_kidx(e, lane)];
  return u.v;
}

__device__ __forceinline__ v16h load_frag_colmaj(const _Float16* M, int ld,
                                                 int k0, int n0, int lane) {
  union { v16h v; _Float16 h[16]; } u;
  int col = n0 + (lane & 15);
#pragma unroll
  for (int e = 0; e < 16; ++e) u.h[e] = M[(k0 + frag_kidx(e, lane)) * ld + col];
  return u.v;
}

// =======================================================================
// K1: per-(b,c) row means (over W) and column means (over H) of x.
//     Coalesced stage to LDS, then LDS-resident reductions.
// =======================================================================
__global__ __launch_bounds__(256) void mss_rowcol_mean(const float* __restrict__ x,
                                                       float* __restrict__ xh,
                                                       float* __restrict__ xw) {
  __shared__ __align__(16) float sm[kHW];
  int bc = blockIdx.x, t = threadIdx.x;
  const float* p = x + (size_t)bc * kHW;
  for (int idx = t; idx < kHW; idx += 256) sm[idx] = p[idx];
  __syncthreads();
  if (t < kH) {
    float rs = 0.f, cs = 0.f;
    for (int i = 0; i < kW; ++i) { rs += sm[t * kW + i]; cs += sm[i * kW + t]; }
    xh[(size_t)bc * kH + t] = rs * (1.f / kW);
    xw[(size_t)bc * kW + t] = cs * (1.f / kH);
  }
}

// =======================================================================
// K2: strip gate: depthwise conv1d (k in {3,5,7,9} per quarter) +
//     4-group GroupNorm + sigmoid.  One block per (b, group, path).
// =======================================================================
__global__ __launch_bounds__(256) void mss_strip_gate(
    const float* __restrict__ xh, const float* __restrict__ xw,
    const float* w3, const float* b3, const float* w5, const float* b5,
    const float* w7, const float* b7, const float* w9, const float* b9,
    const float* gh, const float* bh, const float* gw, const float* bw,
    float* __restrict__ hattn, float* __restrict__ wattn) {
  int id = blockIdx.x;                 // b*8 + grp*2 + path
  int b = id >> 3, grp = (id >> 1) & 3, path = id & 1;
  const float* src = path ? xw : xh;
  const float* wt; const float* bs; int k;
  switch (grp) {
    case 0:  wt = w3; bs = b3; k = 3; break;
    case 1:  wt = w5; bs = b5; k = 5; break;
    case 2:  wt = w7; bs = b7; k = 7; break;
    default: wt = w9; bs = b9; k = 9; break;
  }
  const float* gam = path ? gw : gh;
  const float* bet = path ? bw : bh;
  float* dst = path ? wattn : hattn;

  __shared__ float vals[64 * kH];
  __shared__ float red[256], red2[256];
  int t = threadIdx.x, half = k >> 1;
  const int NV = 64 * kH;

  for (int idx = t; idx < NV; idx += 256) {
    int j = idx / kH, pos = idx % kH;
    const float* row = src + ((size_t)(b * kC + grp * 64 + j)) * kH;
    float s = bs[j];
    for (int tap = 0; tap < k; ++tap) {
      int p2 = pos + tap - half;
      if (p2 >= 0 && p2 < kH) s += wt[j * k + tap] * row[p2];
    }
    vals[idx] = s;
  }
  __syncthreads();
  float ls = 0.f, ls2 = 0.f;
  for (int idx = t; idx < NV; idx += 256) { float v = vals[idx]; ls += v; ls2 += v * v; }
  red[t] = ls; red2[t] = ls2;
  __syncthreads();
  for (int s = 128; s > 0; s >>= 1) {
    if (t < s) { red[t] += red[t + s]; red2[t] += red2[t + s]; }
    __syncthreads();
  }
  float m = red[0] / NV;
  float var = red2[0] / NV - m * m;
  float rs = rsqrtf(var + kEps);
  for (int idx = t; idx < NV; idx += 256) {
    int j = idx / kH, pos = idx % kH, c = grp * 64 + j;
    float v = (vals[idx] - m) * rs * gam[c] + bet[c];
    dst[((size_t)(b * kC + c)) * kH + pos] = 1.f / (1.f + __expf(-v));
  }
}

// =======================================================================
// K3: xs = x * h_attn * w_attn (coalesced), 7x7 window means via LDS
// =======================================================================
__global__ __launch_bounds__(256) void mss_modulate_pool(
    const float* __restrict__ x, const float* __restrict__ hattn,
    const float* __restrict__ wattn, float* __restrict__ xs,
    float* __restrict__ ypool) {
  __shared__ __align__(16) float sm[kHW];
  int bc = blockIdx.x, t = threadIdx.x;
  const float* hp = hattn + (size_t)bc * kH;
  const float* wp = wattn + (size_t)bc * kW;
  const float* p = x + (size_t)bc * kHW;
  float* q = xs + (size_t)bc * kHW;
  for (int idx = t; idx < kHW; idx += 256) {
    int h = idx / kW, w = idx % kW;
    float v = p[idx] * hp[h] * wp[w];
    sm[idx] = v;
    q[idx] = v;
  }
  __syncthreads();
  if (t < kL) {
    int wy = t >> 3, wx = t & 7;
    float s = 0.f;
    for (int dy = 0; dy < kWin; ++dy)
      for (int dx = 0; dx < kWin; ++dx)
        s += sm[(wy * kWin + dy) * kW + wx * kWin + dx];
    ypool[(size_t)bc * kL + t] = s * (1.f / 49.f);
  }
}

// =======================================================================
// K4: 1-group GroupNorm over pooled y per batch (in place)
// =======================================================================
__global__ __launch_bounds__(256) void mss_norm_y(float* __restrict__ ypool,
                                                  const float* __restrict__ g,
                                                  const float* __restrict__ bb) {
  int b = blockIdx.x, t = threadIdx.x;
  float* p = ypool + (size_t)b * kC * kL;
  const int NV = kC * kL;
  __shared__ float red[256], red2[256];
  float ls = 0.f, ls2 = 0.f;
  for (int i = t; i < NV; i += 256) { float v = p[i]; ls += v; ls2 += v * v; }
  red[t] = ls; red2[t] = ls2;
  __syncthreads();
  for (int s = 128; s > 0; s >>= 1) {
    if (t < s) { red[t] += red[t + s]; red2[t] += red2[t + s]; }
    __syncthreads();
  }
  float m = red[0] / NV, var = red2[0] / NV - m * m;
  float rs = rsqrtf(var + kEps);
  for (int i = t; i < NV; i += 256) {
    int c = i >> 6;
    p[i] = (p[i] - m) * rs * g[c] + bb[c];
  }
}

// =======================================================================
// K5: pooled attention via V_WMMA_F32_16X16X32_F16. One wave per (b,head).
// =======================================================================
__global__ __launch_bounds__(32) void mss_attention(
    const float* __restrict__ yn, const float* __restrict__ qw,
    const float* __restrict__ kw, const float* __restrict__ vw,
    float* __restrict__ gate) {
  int b = blockIdx.x / kHeads, hh = blockIdx.x % kHeads;
  int lane = threadIdx.x;

  __shared__ _Float16 qh[kHD * kL], kh[kHD * kL], vh[kHD * kL];
  __shared__ float S[kHD * 33];
  __shared__ _Float16 ah[kHD * kHD];
  __shared__ float O[kHD * kL];

  for (int idx = lane; idx < kHD * kL; idx += 32) {
    int d = idx >> 6, l = idx & 63;
    int c = hh * kHD + d;
    float v = yn[((size_t)(b * kC + c)) * kL + l];
    qh[idx] = (_Float16)(v * qw[c]);
    kh[idx] = (_Float16)(v * kw[c]);
    vh[idx] = (_Float16)(v * vw[c]);
  }
  __syncthreads();

  const float scale = 0.17677669529663687f;  // 32^-0.5
  // ---- S = Q K^T ----
  for (int mi = 0; mi < 2; ++mi)
    for (int ni = 0; ni < 2; ++ni) {
      v8f acc = {};
      for (int k0 = 0; k0 < kL; k0 += 32) {
        v16h a  = load_frag_rowmaj(qh, kL, mi * 16, k0, lane);
        v16h bf = load_frag_rowmaj(kh, kL, ni * 16, k0, lane);  // B(kk,n)=K[n][kk]
        acc = __builtin_amdgcn_wmma_f32_16x16x32_f16(false, a, false, bf,
                                                     (short)0, acc, false, false);
      }
      union { v8f v; float f[8]; } u; u.v = acc;
      int col = ni * 16 + (lane & 15);
      int rbase = mi * 16 + ((lane >> 4) << 3);
#pragma unroll
      for (int r = 0; r < 8; ++r) S[(rbase + r) * 33 + col] = u.f[r] * scale;
    }
  __syncthreads();

  // ---- row softmax (row = lane) ----
  {
    float m = -1e30f;
    for (int e = 0; e < kHD; ++e) m = fmaxf(m, S[lane * 33 + e]);
    float ex[kHD], s = 0.f;
    for (int e = 0; e < kHD; ++e) { ex[e] = __expf(S[lane * 33 + e] - m); s += ex[e]; }
    float inv = 1.f / s;
    for (int e = 0; e < kHD; ++e) ah[lane * kHD + e] = (_Float16)(ex[e] * inv);
  }
  __syncthreads();

  // ---- O = P V ----
  for (int mi = 0; mi < 2; ++mi)
    for (int ni = 0; ni < 4; ++ni) {
      v16h a  = load_frag_rowmaj(ah, kHD, mi * 16, 0, lane);
      v16h bf = load_frag_colmaj(vh, kL, 0, ni * 16, lane);    // B(kk,n)=V[kk][n0+n]
      v8f acc = {};
      acc = __builtin_amdgcn_wmma_f32_16x16x32_f16(false, a, false, bf,
                                                   (short)0, acc, false, false);
      union { v8f v; float f[8]; } u; u.v = acc;
      int col = ni * 16 + (lane & 15);
      int rbase = mi * 16 + ((lane >> 4) << 3);
#pragma unroll
      for (int r = 0; r < 8; ++r) O[(rbase + r) * kL + col] = u.f[r];
    }
  __syncthreads();

  // ---- channel gate ----
  {
    float s = 0.f;
    for (int l = 0; l < kL; ++l) s += O[lane * kL + l];
    s *= (1.f / kL);
    gate[b * kC + hh * kHD + lane] = 1.f / (1.f + __expf(-s));
  }
}

// =======================================================================
// K6: xs *= gate (coalesced in-place), row means + GAP via LDS staging
// =======================================================================
__global__ __launch_bounds__(256) void mss_gate_stats(float* __restrict__ xs,
                                                      const float* __restrict__ gate,
                                                      float* __restrict__ gaph,
                                                      float* __restrict__ gap) {
  __shared__ __align__(16) float sm[kHW];
  __shared__ float red[256];
  int bc = blockIdx.x, t = threadIdx.x;
  float g = gate[bc];
  float* p = xs + (size_t)bc * kHW;
  float part = 0.f;
  for (int idx = t; idx < kHW; idx += 256) {
    float v = p[idx] * g;
    p[idx] = v;
    sm[idx] = v;
    part += v;
  }
  red[t] = part;
  __syncthreads();
  for (int s = 128; s > 0; s >>= 1) { if (t < s) red[t] += red[t + s]; __syncthreads(); }
  if (t == 0) gap[bc] = red[0] * (1.f / kHW);
  if (t < kH) {
    float rs = 0.f;
    for (int w = 0; w < kW; ++w) rs += sm[t * kW + w];
    gaph[(size_t)bc * kH + t] = rs * (1.f / kW);
  }
}

// =======================================================================
// K7: filter coefficients from GAP(xs): f_h (8x3), f_df (8x9 with BN fold)
// =======================================================================
__global__ __launch_bounds__(128) void mss_make_filters(
    const float* __restrict__ gap, const float* __restrict__ shw,
    const float* __restrict__ dfw, const float* __restrict__ bng,
    const float* __restrict__ bnb, float* __restrict__ fh,
    float* __restrict__ fdf) {
  int b = blockIdx.x, t = threadIdx.x;
  const float* g = gap + (size_t)b * kC;
  if (t < 24) {
    float s = 0.f;
    for (int c = 0; c < kC; ++c) s += g[c] * shw[t * kC + c];
    fh[b * 24 + t] = tanhf(s);
  } else if (t < 24 + 72) {
    int i = t - 24;
    float s = 0.f;
    for (int c = 0; c < kC; ++c) s += g[c] * dfw[i * kC + c];
    s = s * (bng[i] * 0.99999500003749968f) + bnb[i];  // / sqrt(1+1e-5)
    fdf[b * 72 + i] = tanhf(s);
  }
}

// =======================================================================
// K8: horizontal strip attention -> o1.  Async-stage xs plane into LDS,
//     compute coalesced, column means & GAP from LDS copy of o1.
// =======================================================================
__global__ __launch_bounds__(256) void mss_hstrip(
    const float* __restrict__ xs, const float* __restrict__ gaph,
    const float* __restrict__ fh, const float* __restrict__ inside,
    const float* __restrict__ lamb_l, const float* __restrict__ lamb_h,
    float* __restrict__ o1, float* __restrict__ gapv, float* __restrict__ gapo) {
  __shared__ __align__(16) float sx[kHW];
  __shared__ __align__(16) float so[kHW];
  __shared__ float red[256];
  int bc = blockIdx.x, b = bc / kC, c = bc % kC, t = threadIdx.x;
  int g = c / kCPG;
  float f0 = fh[b * 24 + g * 3], f1 = fh[b * 24 + g * 3 + 1], f2 = fh[b * 24 + g * 3 + 2];
  float ins = inside[c], ll = lamb_l[c], lh = lamb_h[c] + 1.f;
  const float* p = xs + (size_t)bc * kHW;
  float* q = o1 + (size_t)bc * kHW;

  unsigned sx_base = lds_off_of(sx);
  for (int v = t; v < kHW / 4; v += 256)             // 784 16B chunks
    async_load_b128(sx_base + v * 16u, p + v * 4);
  wait_async0();
  __syncthreads();

  float part = 0.f;
  for (int idx = t; idx < kHW; idx += 256) {
    int h = idx / kW, w = idx % kW;
    int wl = (w == 0) ? 1 : w - 1;
    int wr = (w == kW - 1) ? kW - 2 : w + 1;         // reflect pad
    float out = f0 * sx[h * kW + wl] + f1 * sx[idx] + f2 * sx[h * kW + wr];
    float v = (out * (ins + 1.f) - ins * gaph[(size_t)bc * kH + h]) * ll + sx[idx] * lh;
    so[idx] = v;
    q[idx] = v;
    part += v;
  }
  red[t] = part;
  __syncthreads();
  for (int s = 128; s > 0; s >>= 1) { if (t < s) red[t] += red[t + s]; __syncthreads(); }
  if (t == 0) gapo[bc] = red[0] * (1.f / kHW);
  if (t < kW) {
    float cs = 0.f;
    for (int i = 0; i < kH; ++i) cs += so[i * kW + t];
    gapv[(size_t)bc * kW + t] = cs * (1.f / kH);
  }
}

// =======================================================================
// K9: f_w = tanh(GAP(o1) @ sw_conv_w^T)
// =======================================================================
__global__ __launch_bounds__(32) void mss_make_fw(const float* __restrict__ gapo,
                                                  const float* __restrict__ sww,
                                                  float* __restrict__ fw) {
  int b = blockIdx.x, t = threadIdx.x;
  if (t < 24) {
    const float* g = gapo + (size_t)b * kC;
    float s = 0.f;
    for (int c = 0; c < kC; ++c) s += g[c] * sww[t * kC + c];
    fw[b * 24 + t] = tanhf(s);
  }
}

// =======================================================================
// K10: fused vertical strip (on o1) + cubic combine + dynamic 3x3 filter.
//      Both planes async-staged into LDS; all 12 taps read from LDS.
// =======================================================================
__global__ __launch_bounds__(256) void mss_final(
    const float* __restrict__ xs, const float* __restrict__ o1,
    const float* __restrict__ gap, const float* __restrict__ gapv,
    const float* __restrict__ fw, const float* __restrict__ fdf,
    const float* __restrict__ sw_inside, const float* __restrict__ sw_ll,
    const float* __restrict__ sw_lh, const float* __restrict__ df_inside,
    const float* __restrict__ df_ll, const float* __restrict__ df_lh,
    const float* __restrict__ gamma, const float* __restrict__ beta,
    float* __restrict__ out) {
  __shared__ __align__(16) float sx[kHW];
  __shared__ __align__(16) float so[kHW];
  int bc = blockIdx.x, b = bc / kC, c = bc % kC, t = threadIdx.x;
  int g = c / kCPG;
  float w0 = fw[b * 24 + g * 3], w1 = fw[b * 24 + g * 3 + 1], w2 = fw[b * 24 + g * 3 + 2];
  float d0 = fdf[b * 72 + g * 9 + 0], d1 = fdf[b * 72 + g * 9 + 1], d2 = fdf[b * 72 + g * 9 + 2];
  float d3 = fdf[b * 72 + g * 9 + 3], d4 = fdf[b * 72 + g * 9 + 4], d5 = fdf[b * 72 + g * 9 + 5];
  float d6 = fdf[b * 72 + g * 9 + 6], d7 = fdf[b * 72 + g * 9 + 7], d8 = fdf[b * 72 + g * 9 + 8];
  float insw = sw_inside[c], llw = sw_ll[c], lhw = sw_lh[c] + 1.f;
  float insd = df_inside[c], lld = df_ll[c], lhd = df_lh[c] + 1.f;
  float gm = gamma[c], bt = beta[c], gp = gap[bc];
  const float* xp = xs + (size_t)bc * kHW;
  const float* op = o1 + (size_t)bc * kHW;
  float* dst = out + (size_t)bc * kHW;

  unsigned sx_base = lds_off_of(sx);
  unsigned so_base = lds_off_of(so);
  for (int v = t; v < kHW / 4; v += 256) {           // 784 16B chunks each
    async_load_b128(sx_base + v * 16u, xp + v * 4);
    async_load_b128(so_base + v * 16u, op + v * 4);
  }
  wait_async0();
  __syncthreads();

  for (int idx = t; idx < kHW; idx += 256) {
    int h = idx / kW, w = idx % kW;
    int hu = (h == 0) ? 1 : h - 1;
    int hd = (h == kH - 1) ? kH - 2 : h + 1;         // reflect pad
    int wl = (w == 0) ? 1 : w - 1;
    int wr = (w == kW - 1) ? kW - 2 : w + 1;

    // vertical strip on o1
    float ov = w0 * so[hu * kW + w] + w1 * so[idx] + w2 * so[hd * kW + w];
    float o2 = (ov * (insw + 1.f) - insw * gapv[(size_t)bc * kW + w]) * llw + so[idx] * lhw;
    float x1 = gm * o2 + bt * sx[idx];

    // dynamic 3x3 filter on xs
    float low = d0 * sx[hu * kW + wl] + d1 * sx[hu * kW + w] + d2 * sx[hu * kW + wr] +
                d3 * sx[h  * kW + wl] + d4 * sx[idx]          + d5 * sx[h  * kW + wr] +
                d6 * sx[hd * kW + wl] + d7 * sx[hd * kW + w]  + d8 * sx[hd * kW + wr];
    float x2 = (low * (insd + 1.f) - insd * gp) * lld + sx[idx] * lhd;

    dst[idx] = x1 + x2;
  }
}

// =======================================================================
// launch
// =======================================================================
extern "C" void kernel_launch(void* const* d_in, const int* in_sizes, int n_in,
                              void* d_out, int out_size, void* d_ws, size_t ws_size,
                              hipStream_t stream) {
  (void)in_sizes; (void)n_in; (void)out_size; (void)ws_size;
  const float* x        = (const float*)d_in[0];
  const float* dwc0_w   = (const float*)d_in[1];
  const float* dwc0_b   = (const float*)d_in[2];
  const float* dwc1_w   = (const float*)d_in[3];
  const float* dwc1_b   = (const float*)d_in[4];
  const float* dwc2_w   = (const float*)d_in[5];
  const float* dwc2_b   = (const float*)d_in[6];
  const float* dwc3_w   = (const float*)d_in[7];
  const float* dwc3_b   = (const float*)d_in[8];
  const float* norm_h_g = (const float*)d_in[9];
  const float* norm_h_b = (const float*)d_in[10];
  const float* norm_w_g = (const float*)d_in[11];
  const float* norm_w_b = (const float*)d_in[12];
  const float* norm_g   = (const float*)d_in[13];
  const float* norm_b   = (const float*)d_in[14];
  const float* q_w      = (const float*)d_in[15];
  const float* k_w      = (const float*)d_in[16];
  const float* v_w      = (const float*)d_in[17];
  const float* df_conv_w= (const float*)d_in[18];
  const float* df_bn_g  = (const float*)d_in[19];
  const float* df_bn_b  = (const float*)d_in[20];
  const float* df_lamb_l= (const float*)d_in[21];
  const float* df_lamb_h= (const float*)d_in[22];
  const float* df_inside= (const float*)d_in[23];
  const float* sh_conv_w= (const float*)d_in[24];
  const float* sh_inside= (const float*)d_in[25];
  const float* sh_lamb_l= (const float*)d_in[26];
  const float* sh_lamb_h= (const float*)d_in[27];
  const float* sw_conv_w= (const float*)d_in[28];
  const float* sw_inside= (const float*)d_in[29];
  const float* sw_lamb_l= (const float*)d_in[30];
  const float* sw_lamb_h= (const float*)d_in[31];
  const float* cubic_g  = (const float*)d_in[32];
  const float* cubic_b  = (const float*)d_in[33];
  float* out = (float*)d_out;

  float* ws = (float*)d_ws;
  const size_t N = (size_t)kB * kC * kHW;
  size_t cur = 0;
  auto take = [&](size_t n) { float* p = ws + cur; cur += n; return p; };
  float* xs    = take(N);
  float* o1    = take(N);
  float* xh    = take((size_t)kB * kC * kH);
  float* xw    = take((size_t)kB * kC * kW);
  float* hattn = take((size_t)kB * kC * kH);
  float* wattn = take((size_t)kB * kC * kW);
  float* ypool = take((size_t)kB * kC * kL);
  float* gate  = take((size_t)kB * kC);
  float* gaph  = take((size_t)kB * kC * kH);
  float* gap   = take((size_t)kB * kC);
  float* gapv  = take((size_t)kB * kC * kW);
  float* gapo  = take((size_t)kB * kC);
  float* fh    = take((size_t)kB * 24);
  float* fwv   = take((size_t)kB * 24);
  float* fdf   = take((size_t)kB * 72);

  const int BC = kB * kC;
  mss_rowcol_mean<<<BC, 256, 0, stream>>>(x, xh, xw);
  mss_strip_gate<<<kB * 8, 256, 0, stream>>>(xh, xw,
      dwc0_w, dwc0_b, dwc1_w, dwc1_b, dwc2_w, dwc2_b, dwc3_w, dwc3_b,
      norm_h_g, norm_h_b, norm_w_g, norm_w_b, hattn, wattn);
  mss_modulate_pool<<<BC, 256, 0, stream>>>(x, hattn, wattn, xs, ypool);
  mss_norm_y<<<kB, 256, 0, stream>>>(ypool, norm_g, norm_b);
  mss_attention<<<kB * kHeads, 32, 0, stream>>>(ypool, q_w, k_w, v_w, gate);
  mss_gate_stats<<<BC, 256, 0, stream>>>(xs, gate, gaph, gap);
  mss_make_filters<<<kB, 128, 0, stream>>>(gap, sh_conv_w, df_conv_w,
                                           df_bn_g, df_bn_b, fh, fdf);
  mss_hstrip<<<BC, 256, 0, stream>>>(xs, gaph, fh, sh_inside, sh_lamb_l,
                                     sh_lamb_h, o1, gapv, gapo);
  mss_make_fw<<<kB, 32, 0, stream>>>(gapo, sw_conv_w, fwv);
  mss_final<<<BC, 256, 0, stream>>>(xs, o1, gap, gapv, fwv, fdf,
                                    sw_inside, sw_lamb_l, sw_lamb_h,
                                    df_inside, df_lamb_l, df_lamb_h,
                                    cubic_g, cubic_b, out);
}
